// Pointnet2MSG_54606214201519
// MI455X (gfx1250) — compile-verified, matrix operand order checked by reading, not run on hardware
//
#include <hip/hip_runtime.h>
#include <hip/hip_bf16.h>

typedef __attribute__((ext_vector_type(16))) _Float16     v16h;
typedef __attribute__((ext_vector_type(8)))  float        v8f;
typedef __attribute__((ext_vector_type(4)))  unsigned int v4u;
typedef __attribute__((ext_vector_type(8)))  int          v8i;
typedef __attribute__((ext_vector_type(4)))  int          v4i;

#define DEV __device__ __forceinline__

static constexpr int B = 16;
static constexpr int N = 8192;

#if defined(__has_builtin)
#if __has_builtin(__builtin_amdgcn_tensor_load_to_lds)
#define HAVE_TDM 1
#else
#define HAVE_TDM 0
#endif
#else
#define HAVE_TDM 0
#endif

// ---------------------------------------------------------------------------
// WMMA helpers (CDNA5, wave32): D = A(16x32 f16) x B(32x16 f16) + C(16x16 f32)
// A layout: lane holds row m=lane&15; K halves at kb=(lane>>4)*8 and kb+16.
// B layout (symmetric): lane holds col n=lane&15; same K striping.
// C/D layout: elem i -> (row = i + 8*(lane>>4), col = lane&15).
// ---------------------------------------------------------------------------
DEV v8f wmma16(v16h a, v16h b, v8f c) {
  return __builtin_amdgcn_wmma_f32_16x16x32_f16(false, a, false, b, (short)0, c,
                                                false, false);
}

DEV v16h load_b_frag(const _Float16* __restrict__ W, int C, int kt, int nt, int lane) {
  const int col = nt * 16 + (lane & 15);
  const int kb  = kt * 32 + ((lane >> 4) << 3);
  v16h r;
#pragma unroll
  for (int e = 0; e < 8; ++e) r[e] = W[(size_t)(kb + e) * C + col];
#pragma unroll
  for (int e = 0; e < 8; ++e) r[8 + e] = W[(size_t)(kb + 16 + e) * C + col];
  return r;
}

// B fragment from an LDS-staged column slab Ws[k][16] (k-major, 16 cols)
DEV v16h load_b_frag_lds(const _Float16* Ws, int kt, int lane) {
  const int c  = lane & 15;
  const int kb = kt * 32 + ((lane >> 4) << 3);
  v16h r;
#pragma unroll
  for (int e = 0; e < 8; ++e) r[e] = Ws[(kb + e) * 16 + c];
#pragma unroll
  for (int e = 0; e < 8; ++e) r[8 + e] = Ws[(kb + 16 + e) * 16 + c];
  return r;
}

DEV v16h load_a_rowmajor(const _Float16* __restrict__ X, int ld, int row0, int kt, int lane) {
  const int row = row0 + (lane & 15);
  const _Float16* p = X + (size_t)row * ld + kt * 32 + ((lane >> 4) << 3);
  v16h r;
#pragma unroll
  for (int e = 0; e < 8; ++e) r[e] = p[e];
#pragma unroll
  for (int e = 0; e < 8; ++e) r[8 + e] = p[16 + e];
  return r;
}

// ---------------------------------------------------------------------------
// TDM: stage a KPAD x 16 column slab of W (row-major KPAD x COUT, f16) into LDS.
// Builds a 2D tensor descriptor: tile_dim0 = 16 contiguous columns,
// tile_dim1 = KPAD rows, dim0 stride = COUT elements, data_size = 2 bytes.
// ---------------------------------------------------------------------------
DEV void tdm_stage_wtile(const _Float16* Wcol, _Float16* Ws, int KPAD, int COUT) {
#if HAVE_TDM
  if (threadIdx.x < 32) {
    const unsigned long long ga  = (unsigned long long)(const void*)Wcol;
    const unsigned int       lds = (unsigned int)(unsigned long long)(void*)Ws;
    v4u g0;
    g0[0] = 1u;                                            // count=1, user mode
    g0[1] = lds;                                           // lds_addr
    g0[2] = (unsigned int)(ga & 0xffffffffu);              // global_addr[31:0]
    g0[3] = (unsigned int)((ga >> 32) & 0x01ffffffu)       // global_addr[56:32]
            | 0x80000000u;                                 // type=2 ("image")
    v8i g1;
    g1[0] = 0x00010000;                                    // data_size=2B, no multicast
    g1[1] = 16 << 16;                                      // tensor_dim0 = 16
    g1[2] = (KPAD & 0xffff) << 16;                         // tensor_dim1 = KPAD
    g1[3] = 16 << 16;                                      // tile_dim0 = 16
    g1[4] = KPAD;                                          // tile_dim1 = KPAD
    g1[5] = COUT;                                          // tensor_dim0_stride
    g1[6] = 0;
    g1[7] = 0;
    v4i g2 = {0, 0, 0, 0}, g3 = {0, 0, 0, 0};              // 2D: groups 2/3 unused
#if __clang_major__ >= 23
    v8i g4 = {0, 0, 0, 0, 0, 0, 0, 0};
    __builtin_amdgcn_tensor_load_to_lds(g0, g1, g2, g3, g4, 0);
#else
    __builtin_amdgcn_tensor_load_to_lds(g0, g1, g2, g3, 0);
#endif
    __builtin_amdgcn_s_wait_tensorcnt((short)0);
  }
#else
  for (int i = threadIdx.x; i < KPAD * 16; i += 128)
    Ws[i] = Wcol[(size_t)(i >> 4) * COUT + (i & 15)];
#endif
  __syncthreads();
}

// ---------------------------------------------------------------------------
// 1) split pointcloud into xyz (f32) and feats (f16)
// ---------------------------------------------------------------------------
__global__ void split_kernel(const float* __restrict__ pc,
                             float* __restrict__ xyz, _Float16* __restrict__ f0) {
  const int i = blockIdx.x * blockDim.x + threadIdx.x;
  if (i >= B * N) return;
#pragma unroll
  for (int c = 0; c < 3; ++c) {
    xyz[(size_t)i * 3 + c] = pc[(size_t)i * 6 + c];
    f0[(size_t)i * 3 + c]  = (_Float16)pc[(size_t)i * 6 + 3 + c];
  }
}

// ---------------------------------------------------------------------------
// 2) convert one weight matrix (K x C, f32) to K-padded f16 image
// ---------------------------------------------------------------------------
__global__ void cvt_weight_kernel(const float* __restrict__ W, _Float16* __restrict__ Wh,
                                  int K, int Kpad, int C) {
  const int i = blockIdx.x * blockDim.x + threadIdx.x;
  if (i >= Kpad * C) return;
  const int k = i / C, c = i % C;
  Wh[i] = (k < K) ? (_Float16)W[(size_t)k * C + c] : (_Float16)0.f;
}

// ---------------------------------------------------------------------------
// 3) farthest point sampling; one block per batch; dist array lives in LDS
// ---------------------------------------------------------------------------
__global__ void __launch_bounds__(256) fps_kernel(const float* __restrict__ xyz, int n, int npoint,
                                                  int* __restrict__ idx, float* __restrict__ nxyz) {
  __shared__ float dist[8192];
  __shared__ float rv[256];
  __shared__ int   ri[256];
  const int b = blockIdx.x, tid = threadIdx.x;
  const float* px = xyz + (size_t)b * n * 3;
  for (int i = tid; i < n; i += 256) dist[i] = 1e10f;
  __syncthreads();
  int far = 0;
  for (int it = 0; it < npoint; ++it) {
    if (tid == 0) {
      idx[b * npoint + it] = far;
      nxyz[((size_t)b * npoint + it) * 3 + 0] = px[far * 3 + 0];
      nxyz[((size_t)b * npoint + it) * 3 + 1] = px[far * 3 + 1];
      nxyz[((size_t)b * npoint + it) * 3 + 2] = px[far * 3 + 2];
    }
    const float cx = px[far * 3 + 0], cy = px[far * 3 + 1], cz = px[far * 3 + 2];
    float best = -1.f; int bi = 0;
    for (int i = tid; i < n; i += 256) {
      const float dx = px[i * 3 + 0] - cx, dy = px[i * 3 + 1] - cy, dz = px[i * 3 + 2] - cz;
      const float d  = dx * dx + dy * dy + dz * dz;
      const float nd = fminf(dist[i], d);
      dist[i] = nd;
      if (nd > best) { best = nd; bi = i; }
    }
    rv[tid] = best; ri[tid] = bi;
    __syncthreads();
    for (int off = 128; off > 0; off >>= 1) {
      if (tid < off) {
        if (rv[tid + off] > rv[tid] ||
            (rv[tid + off] == rv[tid] && ri[tid + off] < ri[tid])) {
          rv[tid] = rv[tid + off]; ri[tid] = ri[tid + off];
        }
      }
      __syncthreads();
    }
    far = ri[0];
    __syncthreads();
  }
}

// ---------------------------------------------------------------------------
// 4) ball query: first ns indices with d2 <= r2 (matches sort-then-take)
// ---------------------------------------------------------------------------
__global__ void ball_query_kernel(const float* __restrict__ xyz, const float* __restrict__ nxyz,
                                  int* __restrict__ idx, float r2, int ns, int n, int S) {
  const int t = blockIdx.x * blockDim.x + threadIdx.x;
  if (t >= B * S) return;
  const int b = t / S;
  const float qx = nxyz[(size_t)t * 3 + 0], qy = nxyz[(size_t)t * 3 + 1], qz = nxyz[(size_t)t * 3 + 2];
  const float* px = xyz + (size_t)b * n * 3;
  int* out = idx + (size_t)t * ns;
  int cnt = 0, first = 0;
  for (int i = 0; i < n && cnt < ns; ++i) {
    const float dx = px[i * 3 + 0] - qx, dy = px[i * 3 + 1] - qy, dz = px[i * 3 + 2] - qz;
    if (dx * dx + dy * dy + dz * dz <= r2) {
      if (cnt == 0) first = i;
      out[cnt++] = i;
    }
  }
  if (cnt == 0) { out[0] = 0; cnt = 1; first = 0; }
  for (int j = cnt; j < ns; ++j) out[j] = first;
}

// ---------------------------------------------------------------------------
// 5) fused group MLP + max-pool. One block per (b,s) group, 4 waves.
//    layer1: gather [g_xyz | feats] -> WMMA -> actA (LDS f16)
//    layer2: actA -> WMMA -> actB
//    layer3: actB -> WMMA -> fused column max -> fout
// ---------------------------------------------------------------------------
template <int NS, int K0, int K0PAD, int CF, int C1, int C2, int C3>
__global__ void __launch_bounds__(128) group_mlp_kernel(
    const float* __restrict__ xyz, const _Float16* __restrict__ fin,
    const float* __restrict__ nxyz, const int* __restrict__ idx,
    const _Float16* __restrict__ W1, const float* __restrict__ b1,
    const _Float16* __restrict__ W2, const float* __restrict__ b2,
    const _Float16* __restrict__ W3, const float* __restrict__ b3,
    _Float16* __restrict__ fout, int out_off, int ctot, int S, int n) {
  __shared__ __align__(16) _Float16 actA[NS * C1];
  __shared__ __align__(16) _Float16 actB[NS * C2];

  const int g    = blockIdx.x;           // b*S + s
  const int lane = threadIdx.x & 31;
  const int wave = threadIdx.x >> 5;
  constexpr int MT = NS / 16;

  __builtin_prefetch(W1, 0, 1);          // global_prefetch_b8 into L2 for weights
  __builtin_prefetch(W2, 0, 1);
  __builtin_prefetch(W3, 0, 1);

  const float cx = nxyz[(size_t)g * 3 + 0];
  const float cy = nxyz[(size_t)g * 3 + 1];
  const float cz = nxyz[(size_t)g * 3 + 2];
  const int* gidx  = idx + (size_t)g * NS;
  const int  bbase = (g / S) * n;

  // ---------------- layer 1 (gather fused) ----------------
  for (int mt = wave; mt < MT; mt += 4) {
    const int row = mt * 16 + (lane & 15);
    const int p   = gidx[row];
    const float*    px = xyz + (size_t)(bbase + p) * 3;
    const _Float16* pf = fin + (size_t)(bbase + p) * CF;
    v8f acc[C1 / 16];
#pragma unroll
    for (int nt = 0; nt < C1 / 16; ++nt)
#pragma unroll
      for (int i = 0; i < 8; ++i) acc[nt][i] = 0.f;

#pragma unroll
    for (int kt = 0; kt < K0PAD / 32; ++kt) {
      const int kb = kt * 32 + ((lane >> 4) << 3);
      v16h a;
#pragma unroll
      for (int e = 0; e < 16; ++e) {
        const int k = kb + (e < 8 ? e : 8 + e);   // second half: kb+16..kb+23
        float v = 0.f;
        if (k < 3)       v = px[k] - (k == 0 ? cx : (k == 1 ? cy : cz));
        else if (k < K0) v = (float)pf[k - 3];
        a[e] = (_Float16)v;
      }
#pragma unroll
      for (int nt = 0; nt < C1 / 16; ++nt)
        acc[nt] = wmma16(a, load_b_frag(W1, C1, kt, nt, lane), acc[nt]);
    }
#pragma unroll
    for (int nt = 0; nt < C1 / 16; ++nt) {
      const int col = nt * 16 + (lane & 15);
      const float bv = b1[col];
#pragma unroll
      for (int i = 0; i < 8; ++i) {
        const int r = mt * 16 + i + ((lane >> 4) << 3);
        float v = acc[nt][i] + bv;
        actA[r * C1 + col] = (_Float16)(v > 0.f ? v : 0.f);
      }
    }
  }
  __syncthreads();

  // ---------------- layer 2 ----------------
  for (int mt = wave; mt < MT; mt += 4) {
    v8f acc[C2 / 16];
#pragma unroll
    for (int nt = 0; nt < C2 / 16; ++nt)
#pragma unroll
      for (int i = 0; i < 8; ++i) acc[nt][i] = 0.f;
#pragma unroll
    for (int kt = 0; kt < C1 / 32; ++kt) {
      const v16h a = load_a_rowmajor(actA, C1, mt * 16, kt, lane);
#pragma unroll
      for (int nt = 0; nt < C2 / 16; ++nt)
        acc[nt] = wmma16(a, load_b_frag(W2, C2, kt, nt, lane), acc[nt]);
    }
#pragma unroll
    for (int nt = 0; nt < C2 / 16; ++nt) {
      const int col = nt * 16 + (lane & 15);
      const float bv = b2[col];
#pragma unroll
      for (int i = 0; i < 8; ++i) {
        const int r = mt * 16 + i + ((lane >> 4) << 3);
        float v = acc[nt][i] + bv;
        actB[r * C2 + col] = (_Float16)(v > 0.f ? v : 0.f);
      }
    }
  }
  __syncthreads();

  // ---------------- layer 3 + fused max over group ----------------
  // actA is free now; reuse it as the cross-wave reduction buffer.
  float* red = reinterpret_cast<float*>(actA);
  for (int nt = 0; nt < C3 / 16; ++nt) {
    float cm = 0.f;                       // ReLU outputs are >= 0
    for (int mt = wave; mt < MT; mt += 4) {
      v8f acc;
#pragma unroll
      for (int i = 0; i < 8; ++i) acc[i] = 0.f;
#pragma unroll
      for (int kt = 0; kt < C2 / 32; ++kt)
        acc = wmma16(load_a_rowmajor(actB, C2, mt * 16, kt, lane),
                     load_b_frag(W3, C3, kt, nt, lane), acc);
      const float bv = b3[nt * 16 + (lane & 15)];
#pragma unroll
      for (int i = 0; i < 8; ++i) {
        float v = acc[i] + bv;
        v = v > 0.f ? v : 0.f;
        cm = fmaxf(cm, v);
      }
    }
    cm = fmaxf(cm, __shfl_xor(cm, 16));   // fold the two row-halves (same column)
    if (lane < 16) red[wave * C3 + nt * 16 + lane] = cm;
  }
  __syncthreads();
  for (int c = threadIdx.x; c < C3; c += 128) {
    const float m = fmaxf(fmaxf(red[c], red[C3 + c]),
                          fmaxf(red[2 * C3 + c], red[3 * C3 + c]));
    fout[(size_t)g * ctot + out_off + c] = (_Float16)m;
  }
}

// ---------------------------------------------------------------------------
// 6) SA3 input pack: [xyz2 | feats2] -> 672-padded f16 rows
// ---------------------------------------------------------------------------
__global__ void pack_sa3_kernel(const float* __restrict__ nxyz2,
                                const _Float16* __restrict__ f2, _Float16* __restrict__ X) {
  const int i = blockIdx.x * blockDim.x + threadIdx.x;
  if (i >= B * 128 * 672) return;
  const int row = i / 672, k = i % 672;
  _Float16 v = (_Float16)0.f;
  if (k < 3)        v = (_Float16)nxyz2[(size_t)row * 3 + k];
  else if (k < 643) v = f2[(size_t)row * 640 + (k - 3)];
  X[i] = v;
}

// ---------------------------------------------------------------------------
// 7) WMMA GEMM + ReLU: Y[R x COUT] = relu(X[R x KPAD] @ W + b)
//    Block = 4 waves sharing one N-tile; the KPAD x 16 weight slab is staged
//    into LDS by the Tensor Data Mover, then B-fragments come from LDS.
// ---------------------------------------------------------------------------
template <int KPAD, int COUT>
__global__ void __launch_bounds__(128) gemm_relu_kernel(
    const _Float16* __restrict__ X, const _Float16* __restrict__ W,
    const float* __restrict__ bias, _Float16* __restrict__ Y, int R) {
  __shared__ __align__(16) _Float16 Ws[KPAD * 16];
  const int lane = threadIdx.x & 31, wave = threadIdx.x >> 5;
  constexpr int NT = COUT / 16;
  const int nt  = blockIdx.x % NT;        // all 4 waves share this N-tile
  const int mt  = (blockIdx.x / NT) * 4 + wave;

  tdm_stage_wtile(W + (size_t)nt * 16, Ws, KPAD, COUT);

  const float bv = bias[nt * 16 + (lane & 15)];
  v8f acc;
#pragma unroll
  for (int i = 0; i < 8; ++i) acc[i] = 0.f;
#pragma unroll 4
  for (int kt = 0; kt < KPAD / 32; ++kt)
    acc = wmma16(load_a_rowmajor(X, KPAD, mt * 16, kt, lane),
                 load_b_frag_lds(Ws, kt, lane), acc);
#pragma unroll
  for (int i = 0; i < 8; ++i) {
    const int r = mt * 16 + i + ((lane >> 4) << 3);
    float v = acc[i] + bv;
    v = v > 0.f ? v : 0.f;
    Y[(size_t)r * COUT + nt * 16 + (lane & 15)] = (_Float16)v;
  }
}

// ---------------------------------------------------------------------------
// 8) final max over 128 points per batch -> f32 output
// ---------------------------------------------------------------------------
__global__ void max_rows_kernel(const _Float16* __restrict__ A, float* __restrict__ out) {
  const int t = blockIdx.x * blockDim.x + threadIdx.x;
  if (t >= B * 1024) return;
  const int b = t / 1024, c = t % 1024;
  float m = -1e30f;
  for (int s = 0; s < 128; ++s)
    m = fmaxf(m, (float)A[((size_t)b * 128 + s) * 1024 + c]);
  out[t] = m;
}

// ---------------------------------------------------------------------------
// host side
// ---------------------------------------------------------------------------
extern "C" void kernel_launch(void* const* d_in, const int* in_sizes, int n_in,
                              void* d_out, int out_size, void* d_ws, size_t ws_size,
                              hipStream_t stream) {
  (void)in_sizes; (void)n_in; (void)out_size; (void)ws_size;
  char* base = (char*)d_ws;
  size_t off = 0;
  auto alloc = [&](size_t bytes) -> void* {
    void* p = base + off;
    off += (bytes + 255) & ~(size_t)255;
    return p;
  };

  float*    xyz0   = (float*)alloc((size_t)B * N * 3 * 4);
  _Float16* f0h    = (_Float16*)alloc((size_t)B * N * 3 * 2);
  int*      fidx1  = (int*)alloc((size_t)B * 512 * 4);
  float*    nxyz1  = (float*)alloc((size_t)B * 512 * 3 * 4);
  int*      idx1a  = (int*)alloc((size_t)B * 512 * 16 * 4);
  int*      idx1b  = (int*)alloc((size_t)B * 512 * 32 * 4);
  int*      idx1c  = (int*)alloc((size_t)B * 512 * 128 * 4);
  _Float16* feats1 = (_Float16*)alloc((size_t)B * 512 * 320 * 2);
  int*      fidx2  = (int*)alloc((size_t)B * 128 * 4);
  float*    nxyz2  = (float*)alloc((size_t)B * 128 * 3 * 4);
  int*      idx2a  = (int*)alloc((size_t)B * 128 * 32 * 4);
  int*      idx2b  = (int*)alloc((size_t)B * 128 * 64 * 4);
  int*      idx2c  = (int*)alloc((size_t)B * 128 * 128 * 4);
  _Float16* feats2 = (_Float16*)alloc((size_t)B * 128 * 640 * 2);
  _Float16* X3     = (_Float16*)alloc((size_t)B * 128 * 672 * 2);
  _Float16* A3a    = (_Float16*)alloc((size_t)B * 128 * 256 * 2);
  _Float16* A3b    = (_Float16*)alloc((size_t)B * 128 * 512 * 2);
  _Float16* A3c    = (_Float16*)alloc((size_t)B * 128 * 1024 * 2);

  // 21 MLP layers in jax-tree leaf order: (W,b) pairs, SA1 s0..s2, SA2 s0..s2, SA3
  static const int LK[21]  = {6,32,32, 6,64,64, 6,64,96,
                              323,64,64, 323,128,128, 323,128,128,
                              643,256,512};
  static const int LKP[21] = {32,32,32, 32,64,64, 32,64,96,
                              352,64,64, 352,128,128, 352,128,128,
                              672,256,512};
  static const int LC[21]  = {32,32,64, 64,64,128, 64,96,128,
                              64,64,128, 128,128,256, 128,128,256,
                              256,512,1024};
  _Float16*    Wh[21];
  const float* Wf[21];
  const float* Bf[21];
  for (int l = 0; l < 21; ++l) {
    Wh[l] = (_Float16*)alloc((size_t)LKP[l] * LC[l] * 2);
    Wf[l] = (const float*)d_in[1 + 2 * l];
    Bf[l] = (const float*)d_in[2 + 2 * l];
  }
  const float* pc  = (const float*)d_in[0];
  float*       out = (float*)d_out;

  // 1) split
  split_kernel<<<(B * N + 255) / 256, 256, 0, stream>>>(pc, xyz0, f0h);

  // 2) weight conversion
  for (int l = 0; l < 21; ++l) {
    const int total = LKP[l] * LC[l];
    cvt_weight_kernel<<<(total + 255) / 256, 256, 0, stream>>>(Wf[l], Wh[l], LK[l], LKP[l], LC[l]);
  }

  // 3) SA1: FPS 8192 -> 512
  fps_kernel<<<B, 256, 0, stream>>>(xyz0, N, 512, fidx1, nxyz1);

  // 4) SA1 ball queries (r2 = 0.01, 0.04, 0.16)
  {
    const int g = (B * 512 + 255) / 256;
    ball_query_kernel<<<g, 256, 0, stream>>>(xyz0, nxyz1, idx1a, 0.01f, 16,  N, 512);
    ball_query_kernel<<<g, 256, 0, stream>>>(xyz0, nxyz1, idx1b, 0.04f, 32,  N, 512);
    ball_query_kernel<<<g, 256, 0, stream>>>(xyz0, nxyz1, idx1c, 0.16f, 128, N, 512);
  }

  // 5) SA1 grouped MLPs (output feats1: 64 | 128 | 128 = 320 ch)
  group_mlp_kernel<16, 6, 32, 3, 32, 32, 64><<<B * 512, 128, 0, stream>>>(
      xyz0, f0h, nxyz1, idx1a, Wh[0], Bf[0], Wh[1], Bf[1], Wh[2], Bf[2],
      feats1, 0, 320, 512, N);
  group_mlp_kernel<32, 6, 32, 3, 64, 64, 128><<<B * 512, 128, 0, stream>>>(
      xyz0, f0h, nxyz1, idx1b, Wh[3], Bf[3], Wh[4], Bf[4], Wh[5], Bf[5],
      feats1, 64, 320, 512, N);
  group_mlp_kernel<128, 6, 32, 3, 64, 96, 128><<<B * 512, 128, 0, stream>>>(
      xyz0, f0h, nxyz1, idx1c, Wh[6], Bf[6], Wh[7], Bf[7], Wh[8], Bf[8],
      feats1, 192, 320, 512, N);

  // 6) SA2: FPS 512 -> 128
  fps_kernel<<<B, 256, 0, stream>>>(nxyz1, 512, 128, fidx2, nxyz2);

  // 7) SA2 ball queries (r2 = 0.04, 0.16, 0.64)
  {
    const int g = (B * 128 + 255) / 256;
    ball_query_kernel<<<g, 256, 0, stream>>>(nxyz1, nxyz2, idx2a, 0.04f, 32,  512, 128);
    ball_query_kernel<<<g, 256, 0, stream>>>(nxyz1, nxyz2, idx2b, 0.16f, 64,  512, 128);
    ball_query_kernel<<<g, 256, 0, stream>>>(nxyz1, nxyz2, idx2c, 0.64f, 128, 512, 128);
  }

  // 8) SA2 grouped MLPs (output feats2: 128 | 256 | 256 = 640 ch)
  group_mlp_kernel<32, 323, 352, 320, 64, 64, 128><<<B * 128, 128, 0, stream>>>(
      nxyz1, feats1, nxyz2, idx2a, Wh[9], Bf[9], Wh[10], Bf[10], Wh[11], Bf[11],
      feats2, 0, 640, 128, 512);
  group_mlp_kernel<64, 323, 352, 320, 128, 128, 256><<<B * 128, 128, 0, stream>>>(
      nxyz1, feats1, nxyz2, idx2b, Wh[12], Bf[12], Wh[13], Bf[13], Wh[14], Bf[14],
      feats2, 128, 640, 128, 512);
  group_mlp_kernel<128, 323, 352, 320, 128, 128, 256><<<B * 128, 128, 0, stream>>>(
      nxyz1, feats1, nxyz2, idx2c, Wh[15], Bf[15], Wh[16], Bf[16], Wh[17], Bf[17],
      feats2, 384, 640, 128, 512);

  // 9) SA3 global MLP 643 -> 256 -> 512 -> 1024, then max over 128 points
  pack_sa3_kernel<<<(B * 128 * 672 + 255) / 256, 256, 0, stream>>>(nxyz2, feats2, X3);
  {
    const int R = B * 128;                       // 2048 rows, 128 M-tiles
    gemm_relu_kernel<672, 256><<<(R / 64) * (256 / 16), 128, 0, stream>>>(
        X3, Wh[18], Bf[18], A3a, R);
    gemm_relu_kernel<256, 512><<<(R / 64) * (512 / 16), 128, 0, stream>>>(
        A3a, Wh[19], Bf[19], A3b, R);
    gemm_relu_kernel<512, 1024><<<(R / 64) * (1024 / 16), 128, 0, stream>>>(
        A3b, Wh[20], Bf[20], A3c, R);
  }
  max_rows_kernel<<<(B * 1024 + 255) / 256, 256, 0, stream>>>(A3c, out);
}
